// LowMixer_2714419331333
// MI455X (gfx1250) — matrix operand order repaired and checked by reading, not compiled, not run on hardware
//
#include <hip/hip_runtime.h>

#define CC   512
#define C3   1536
#define CN   3136
#define NHEAD 8
#define HD    64

typedef __attribute__((ext_vector_type(16))) __bf16 v16bf;
typedef __attribute__((ext_vector_type(8)))  float  v8f;
typedef __attribute__((ext_vector_type(2)))  __bf16 bf16x2;
typedef __attribute__((ext_vector_type(4)))  unsigned int u32x4;
typedef __attribute__((ext_vector_type(8)))  int i32x8;
typedef __attribute__((ext_vector_type(4)))  int i32x4;

// ---------------- fragment loaders (CDNA5 wave32 WMMA layouts) ----------------
// A 16x32 bf16: lane L -> M=L%16 ; vgpr v holds K = ((v<4)?0:16) + 8*(L>>4) + 2*(v&3) + {0,1}
// lds layout must be [row][K] row-major (pairs contiguous).
__device__ __forceinline__ v16bf load_a_frag(const __bf16* lds, int row, int stride,
                                             int kbase, int lane) {
  v16bf a;
  const int half = lane >> 4;
  const __bf16* p = lds + row * stride + kbase;
#pragma unroll
  for (int v = 0; v < 8; ++v) {
    int k = ((v < 4) ? 0 : 16) + half * 8 + (v & 3) * 2;
    a[2 * v]     = p[k];
    a[2 * v + 1] = p[k + 1];
  }
  return a;
}

// B 32x16 bf16: lane L -> N=L%16 ; vgpr v holds K = ((v<4)?0:16) + 8*(L>>4) + 2*(v&3) + {0,1}
// lds layout must be [N][K] (pairs contiguous along K).
__device__ __forceinline__ v16bf load_b_frag(const __bf16* lds, int ncol, int stride,
                                             int kbase, int lane) {
  v16bf b;
  const __bf16* p = lds + ncol * stride + kbase + (lane >> 4) * 8;
#pragma unroll
  for (int v = 0; v < 8; ++v) {
    int k = ((v < 4) ? 0 : 16) + (v & 3) * 2;
    b[2 * v]     = p[k];
    b[2 * v + 1] = p[k + 1];
  }
  return b;
}

// reductions across the 16-lane group that owns one D-matrix row
__device__ __forceinline__ float rowmax16(float v) {
#pragma unroll
  for (int m = 8; m >= 1; m >>= 1) v = fmaxf(v, __shfl_xor(v, m, 16));
  return v;
}
__device__ __forceinline__ float rowsum16(float v) {
#pragma unroll
  for (int m = 8; m >= 1; m >>= 1) v += __shfl_xor(v, m, 16);
  return v;
}

// ---------------- K1: fp32 weights -> bf16 ----------------
__global__ void k_conv_w(const float* __restrict__ w, __bf16* __restrict__ wbf, int n) {
  int t = blockIdx.x * blockDim.x + threadIdx.x;
  if (t < n) wbf[t] = (__bf16)w[t];
}

// ---------------- K2: QKV GEMM  Y[b](1536 x 3136) = W(1536x512) @ X[b](512x3136) ----
// grid (49 ntiles, 6 mtiles, 32 batch-slices), 256 threads = 8 waves.
// tile: 256 rows x 64 cols, K-step 32.  A tile staged by the Tensor Data Mover
// (async DMA, wave 0 issues, overlaps with fp32->bf16 conversion of the B tile).
__global__ __launch_bounds__(256) void k_qkv_gemm(const float* __restrict__ x1,
                                                  const float* __restrict__ x2,
                                                  const __bf16* __restrict__ wbf,
                                                  __bf16* __restrict__ qkv) {
  __shared__ __bf16 ldsA[256 * 32];   // [m][k]  (filled by TDM)
  __shared__ __bf16 ldsB[64 * 32];    // [n][k]
  const int n0 = blockIdx.x * 64;
  const int m0 = blockIdx.y * 256;
  const int bb = blockIdx.z;
  const int branch = bb >> 4, b = bb & 15;
  const float* X = (branch ? x2 : x1) + (size_t)b * CC * CN;
  __bf16* Y = qkv + ((size_t)branch * 16 + b) * (size_t)C3 * CN;
  const int tid = threadIdx.x, lane = tid & 31, wv = tid >> 5;
  const unsigned int ldsA_addr = (unsigned int)(size_t)(void*)&ldsA[0];

  v8f acc[2][4] = {};

  for (int k0 = 0; k0 < CC; k0 += 32) {
    __syncthreads();
    if (wv == 0) {
      // TDM descriptor: 2D tile, 32 (K, contiguous) x 256 (M rows), elem=2B,
      // row stride 512 elems.  D# group0/group1 per CDNA5 ISA 8.3/8.4.
      unsigned long long ga =
          (unsigned long long)(size_t)(wbf + (size_t)m0 * CC + k0);
      u32x4 g0;
      g0[0] = 1u;                                        // count=1, user mode
      g0[1] = ldsA_addr;                                 // lds_addr
      g0[2] = (unsigned int)ga;                          // global_addr[31:0]
      g0[3] = (unsigned int)(ga >> 32) | (2u << 30);     // global_addr[56:32] | type=2
      i32x8 g1;
      g1[0] = 0x10000;                 // workgroup_mask=0, data_size=1 (2 bytes)
      g1[1] = (int)(((unsigned)CC & 0xFFFFu) << 16);     // tensor_dim0[15:0] @ bits 63:48
      g1[2] = (int)(((unsigned)C3 & 0xFFFFu) << 16);     // tensor_dim1[15:0] @ bits 95:80
      g1[3] = (int)(32u << 16);        // tile_dim0=32 @ bits 127:112
      g1[4] = 256;                     // tile_dim1=256 @ bits 143:128, tile_dim2=0
      g1[5] = CC;                      // tensor_dim0_stride[31:0] = 512
      g1[6] = 0;
      g1[7] = 0;
      i32x4 z4 = {0, 0, 0, 0};
      i32x8 z8 = {0, 0, 0, 0, 0, 0, 0, 0};
      __builtin_amdgcn_tensor_load_to_lds(g0, g1, z4, z4, z8, 0);
    }
    {   // B tile: fp32 -> bf16, packed k-pair stores, layout [n][k]
      int a2 = tid & 15;               // k-pair: rows k0+2*a2, k0+2*a2+1
      int ng = tid >> 4;               // n sub-group of 4
      const float* r0 = X + (size_t)(k0 + 2 * a2) * CN + n0 + ng * 4;
      const float* r1 = r0 + CN;
      if (k0 + 32 < CC) __builtin_prefetch(r0 + (size_t)32 * CN, 0, 1);
#pragma unroll
      for (int i = 0; i < 4; ++i) {
        bf16x2 p;
        p[0] = (__bf16)r0[i];
        p[1] = (__bf16)r1[i];
        *(bf16x2*)&ldsB[(ng * 4 + i) * 32 + 2 * a2] = p;
      }
    }
    if (wv == 0) __builtin_amdgcn_s_wait_tensorcnt((short)0);
    __syncthreads();
#pragma unroll
    for (int mt = 0; mt < 2; ++mt) {
      v16bf a = load_a_frag(ldsA, wv * 32 + mt * 16 + (lane & 15), 32, 0, lane);
#pragma unroll
      for (int nt = 0; nt < 4; ++nt) {
        v16bf bf = load_b_frag(ldsB, nt * 16 + (lane & 15), 32, 0, lane);
        acc[mt][nt] = __builtin_amdgcn_wmma_f32_16x16x32_bf16(
            false, a, false, bf, (short)0, acc[mt][nt], false, false);
      }
    }
  }
#pragma unroll
  for (int mt = 0; mt < 2; ++mt)
#pragma unroll
    for (int nt = 0; nt < 4; ++nt) {
      int col = n0 + nt * 16 + (lane & 15);
      int row = m0 + wv * 32 + mt * 16 + 8 * (lane >> 4);
#pragma unroll
      for (int r = 0; r < 8; ++r)
        Y[(size_t)(row + r) * CN + col] = (__bf16)acc[mt][nt][r];
    }
}

// ---------------- K3: agent = scale * avgpool8x8(|q1-q2|) ----------------
__global__ void k_pool_agent(const __bf16* __restrict__ qkv, __bf16* __restrict__ agent) {
  int t = blockIdx.x * blockDim.x + threadIdx.x;      // B*64*C
  if (t >= 16 * 64 * CC) return;
  int c = t & (CC - 1), a = (t >> 9) & 63, b = t >> 15;
  int ph = a >> 3, pw = a & 7;
  const __bf16* q1 = qkv + (size_t)b * C3 * CN + (size_t)c * CN;
  const __bf16* q2 = q1 + (size_t)16 * C3 * CN;
  float s = 0.f;
#pragma unroll
  for (int dy = 0; dy < 7; ++dy)
#pragma unroll
    for (int dx = 0; dx < 7; ++dx) {
      int hw = (ph * 7 + dy) * 56 + pw * 7 + dx;
      s += fabsf((float)q1[hw] - (float)q2[hw]);
    }
  agent[t] = (__bf16)(s * (1.0f / 49.0f) * 0.125f);   // scale = hd^-0.5 folded in
}

// ---------------- K4: agent attention (flash softmax over n=3136) ----------------
// one WG per (branch,b,head): S1 = ah@khT -> online softmax -> agent_v = P@vh
__global__ __launch_bounds__(128) void k_agent_attn(const __bf16* __restrict__ qkv,
                                                    const __bf16* __restrict__ agent,
                                                    __bf16* __restrict__ agentv) {
  __shared__ __bf16 ldsAh[64 * 64];   // [a][d]   (A operand)
  __shared__ __bf16 ldsK [64 * 64];   // [j][d]   (B: [N=j][K=d])
  __shared__ __bf16 ldsV [64 * 64];   // [d][j]   (B: [N=d][K=j])
  __shared__ __bf16 ldsP [64 * 64];   // [a][j]   (A re-staging)
  const int gid = blockIdx.x;
  const int branch = gid >> 7, rem = gid & 127, b = rem >> 3, head = rem & 7;
  const __bf16* base = qkv + ((size_t)branch * 16 + b) * (size_t)C3 * CN;
  const __bf16* Kp = base + (size_t)(CC + head * HD) * CN;
  const __bf16* Vp = base + (size_t)(2 * CC + head * HD) * CN;
  const __bf16* Ah = agent + (size_t)b * 64 * CC + head * HD;
  const int tid = threadIdx.x, lane = tid & 31, wv = tid >> 5;

  for (int i = tid; i < 64 * 64; i += 128)
    ldsAh[i] = Ah[(size_t)(i >> 6) * CC + (i & 63)];
  __syncthreads();

  float run_max[8], run_sum[8];
  v8f acc[4] = {};
#pragma unroll
  for (int r = 0; r < 8; ++r) { run_max[r] = -1e30f; run_sum[r] = 0.f; }

  for (int jc = 0; jc < 49; ++jc) {
    const int jb = jc * 64;
    __syncthreads();
    {   // stage K (transposed) and V (as-is) chunks
      int d = tid >> 1, joff = (tid & 1) * 32;
      const __bf16* sk = Kp + (size_t)d * CN + jb + joff;
      const __bf16* sv = Vp + (size_t)d * CN + jb + joff;
#pragma unroll
      for (int i = 0; i < 32; ++i) ldsK[(joff + i) * 64 + d] = sk[i];
#pragma unroll
      for (int i = 0; i < 32; ++i) ldsV[d * 64 + joff + i] = sv[i];
    }
    __syncthreads();

    v8f S[4] = {};
#pragma unroll
    for (int ks = 0; ks < 2; ++ks) {
      v16bf a = load_a_frag(ldsAh, wv * 16 + (lane & 15), 64, ks * 32, lane);
#pragma unroll
      for (int nt = 0; nt < 4; ++nt) {
        v16bf bf = load_b_frag(ldsK, nt * 16 + (lane & 15), 64, ks * 32, lane);
        S[nt] = __builtin_amdgcn_wmma_f32_16x16x32_bf16(
            false, a, false, bf, (short)0, S[nt], false, false);
      }
    }
    // online softmax per row (row = r + 8*(lane>>4), spread over one 16-lane group)
#pragma unroll
    for (int r = 0; r < 8; ++r) {
      float m = fmaxf(fmaxf(S[0][r], S[1][r]), fmaxf(S[2][r], S[3][r]));
      m = rowmax16(m);
      float nm = fmaxf(run_max[r], m);
      float alpha = __expf(run_max[r] - nm);
      run_max[r] = nm;
      float ps = 0.f;
#pragma unroll
      for (int nt = 0; nt < 4; ++nt) {
        float p = __expf(S[nt][r] - nm);
        S[nt][r] = p;
        ps += p;
      }
      ps = rowsum16(ps);
      run_sum[r] = run_sum[r] * alpha + ps;
#pragma unroll
      for (int nt = 0; nt < 4; ++nt) acc[nt][r] *= alpha;
    }
    // re-stage P through per-wave LDS region (same-wave DS ordering suffices)
#pragma unroll
    for (int nt = 0; nt < 4; ++nt)
#pragma unroll
      for (int r = 0; r < 8; ++r)
        ldsP[(wv * 16 + r + 8 * (lane >> 4)) * 64 + nt * 16 + (lane & 15)] =
            (__bf16)S[nt][r];
    // acc += P @ V
#pragma unroll
    for (int ks = 0; ks < 2; ++ks) {
      v16bf a = load_a_frag(ldsP, wv * 16 + (lane & 15), 64, ks * 32, lane);
#pragma unroll
      for (int nt = 0; nt < 4; ++nt) {
        v16bf bf = load_b_frag(ldsV, nt * 16 + (lane & 15), 64, ks * 32, lane);
        acc[nt] = __builtin_amdgcn_wmma_f32_16x16x32_bf16(
            false, a, false, bf, (short)0, acc[nt], false, false);
      }
    }
  }
  __bf16* out = agentv + (((size_t)branch * 16 + b) * NHEAD + head) * 64 * HD;
#pragma unroll
  for (int nt = 0; nt < 4; ++nt) {
    int d = nt * 16 + (lane & 15);
    int ab = wv * 16 + 8 * (lane >> 4);
#pragma unroll
    for (int r = 0; r < 8; ++r)
      out[(ab + r) * HD + d] = (__bf16)(acc[nt][r] / run_sum[r]);
  }
}

// ---------------- K5: pixel attention + fp32 store ----------------
// grid (49 pixel tiles, 256 (branch,b,head)), 128 threads. 64 pixels per WG.
__global__ __launch_bounds__(128) void k_pixel_attn(const __bf16* __restrict__ qkv,
                                                    const __bf16* __restrict__ agent,
                                                    const __bf16* __restrict__ agentv,
                                                    float* __restrict__ out) {
  __shared__ __bf16 ldsAg[64 * 64];   // [a][d]  (B: [N=a][K=d])
  __shared__ __bf16 ldsQ [64 * 64];   // [j][d]  (A)
  __shared__ __bf16 ldsAV[64 * 64];   // [d][a]  (B: [N=d][K=a])
  __shared__ __bf16 ldsP [64 * 64];   // [j][a]  (A)
  const int jb = blockIdx.x * 64;
  const int gid = blockIdx.y;
  const int branch = gid >> 7, rem = gid & 127, b = rem >> 3, head = rem & 7;
  const __bf16* Qp = qkv + ((size_t)branch * 16 + b) * (size_t)C3 * CN +
                     (size_t)(head * HD) * CN;
  const __bf16* Ah = agent + (size_t)b * 64 * CC + head * HD;
  const __bf16* AV = agentv + (((size_t)branch * 16 + b) * NHEAD + head) * 64 * HD;
  float* Op = out + ((size_t)branch * 16 + b) * (size_t)CC * CN +
              (size_t)(head * HD) * CN;
  const int tid = threadIdx.x, lane = tid & 31, wv = tid >> 5;

  for (int i = tid; i < 64 * 64; i += 128) {
    int a = i >> 6, d = i & 63;
    ldsAg[i] = Ah[(size_t)a * CC + d];
    ldsAV[d * 64 + a] = AV[i];
  }
  {
    int d = tid >> 1, joff = (tid & 1) * 32;
    const __bf16* s = Qp + (size_t)d * CN + jb + joff;
#pragma unroll
    for (int i = 0; i < 32; ++i) ldsQ[(joff + i) * 64 + d] = s[i];
  }
  __syncthreads();

  v8f S[4] = {};
#pragma unroll
  for (int ks = 0; ks < 2; ++ks) {
    v16bf a = load_a_frag(ldsQ, wv * 16 + (lane & 15), 64, ks * 32, lane);
#pragma unroll
    for (int nt = 0; nt < 4; ++nt) {
      v16bf bf = load_b_frag(ldsAg, nt * 16 + (lane & 15), 64, ks * 32, lane);
      S[nt] = __builtin_amdgcn_wmma_f32_16x16x32_bf16(
          false, a, false, bf, (short)0, S[nt], false, false);
    }
  }
  // full softmax over the 64 agents
#pragma unroll
  for (int r = 0; r < 8; ++r) {
    float m = fmaxf(fmaxf(S[0][r], S[1][r]), fmaxf(S[2][r], S[3][r]));
    m = rowmax16(m);
    float ps = 0.f;
#pragma unroll
    for (int nt = 0; nt < 4; ++nt) {
      float p = __expf(S[nt][r] - m);
      S[nt][r] = p;
      ps += p;
    }
    ps = rowsum16(ps);
    float inv = 1.0f / ps;
#pragma unroll
    for (int nt = 0; nt < 4; ++nt)
      ldsP[(wv * 16 + r + 8 * (lane >> 4)) * 64 + nt * 16 + (lane & 15)] =
          (__bf16)(S[nt][r] * inv);
  }
  v8f acc[4] = {};
#pragma unroll
  for (int ks = 0; ks < 2; ++ks) {
    v16bf a = load_a_frag(ldsP, wv * 16 + (lane & 15), 64, ks * 32, lane);
#pragma unroll
    for (int nt = 0; nt < 4; ++nt) {
      v16bf bf = load_b_frag(ldsAV, nt * 16 + (lane & 15), 64, ks * 32, lane);
      acc[nt] = __builtin_amdgcn_wmma_f32_16x16x32_bf16(
          false, a, false, bf, (short)0, acc[nt], false, false);
    }
  }
#pragma unroll
  for (int nt = 0; nt < 4; ++nt) {
    int d = nt * 16 + (lane & 15);
    float* dst = Op + (size_t)d * CN + jb + wv * 16 + 8 * (lane >> 4);
#pragma unroll
    for (int r = 0; r < 8; ++r) dst[r] = acc[nt][r];
  }
}

// ---------------- host ----------------
extern "C" void kernel_launch(void* const* d_in, const int* in_sizes, int n_in,
                              void* d_out, int out_size, void* d_ws, size_t ws_size,
                              hipStream_t stream) {
  const float* x1 = (const float*)d_in[0];
  const float* x2 = (const float*)d_in[1];
  const float* qw = (const float*)d_in[2];
  float* out = (float*)d_out;

  size_t off = 0;
  auto carve = [&](size_t bytes) {
    void* p = (char*)d_ws + off;
    off += (bytes + 255) & ~(size_t)255;
    return p;
  };
  __bf16* wbf    = (__bf16*)carve((size_t)C3 * CC * 2);                 // 1.5 MB
  __bf16* qkvb   = (__bf16*)carve((size_t)2 * 16 * C3 * CN * 2);        // 308 MB
  __bf16* agent  = (__bf16*)carve((size_t)16 * 64 * CC * 2);            // 1 MB
  __bf16* agentv = (__bf16*)carve((size_t)2 * 16 * NHEAD * 64 * HD * 2);// 2 MB

  k_conv_w<<<(C3 * CC + 255) / 256, 256, 0, stream>>>(qw, wbf, C3 * CC);
  k_qkv_gemm<<<dim3(CN / 64, C3 / 256, 32), 256, 0, stream>>>(x1, x2, wbf, qkvb);
  k_pool_agent<<<(16 * 64 * CC + 255) / 256, 256, 0, stream>>>(qkvb, agent);
  k_agent_attn<<<256, 128, 0, stream>>>(qkvb, agent, agentv);
  k_pixel_attn<<<dim3(CN / 64, 256), 128, 0, stream>>>(qkvb, agent, agentv, out);
}
